// GenericRNNSequenceEncoder_24257975287916
// MI455X (gfx1250) — matrix operand order repaired and checked by reading, not compiled, not run on hardware
//
#include <hip/hip_runtime.h>
#include <hip/hip_bf16.h>
#include <math.h>

#define B_DIM 64
#define T_LEN 512
#define D_DIM 512
#define H_DIM 512
#define Z_DIM (4 * H_DIM)
#define MROWS (B_DIM * T_LEN)  // 32768 flat rows for the big GEMM

typedef __attribute__((ext_vector_type(16))) __bf16 v16bf;
typedef __attribute__((ext_vector_type(8)))  float  v8f;

struct alignas(16) B128 { unsigned int w[4]; };

union AFrag {
    v16bf v;
    unsigned short s[16];
};

__device__ __forceinline__ unsigned short f2bf(float f) {
    unsigned int u = __float_as_uint(f);
    unsigned int r = u + 0x7FFFu + ((u >> 16) & 1u);  // RNE
    return (unsigned short)(r >> 16);
}

__device__ __forceinline__ void copy16B(unsigned short* dst, const unsigned short* src) {
    *(B128*)dst = *(const B128*)src;
}

__device__ __forceinline__ float sigmoidf_(float x) {
    return 1.0f / (1.0f + __expf(-x));
}

// ---------------------------------------------------------------------------
// Convert Wi/Wh fp32 -> bf16 (row-major [K, 4H]); 4 MB total, L2-resident.
// ---------------------------------------------------------------------------
__global__ void convert_weights_kernel(const float* __restrict__ Wi,
                                       const float* __restrict__ Wh,
                                       unsigned short* __restrict__ Wi_bf,
                                       unsigned short* __restrict__ Wh_bf) {
    const int n = D_DIM * Z_DIM;
    for (int i = blockIdx.x * blockDim.x + threadIdx.x; i < n;
         i += gridDim.x * blockDim.x) {
        Wi_bf[i] = f2bf(Wi[i]);
        Wh_bf[i] = f2bf(Wh[i]);
    }
}

// ---------------------------------------------------------------------------
// Convert inputs fp32 -> bf16 once (removes VALU conversions from the scan).
// ---------------------------------------------------------------------------
__global__ void convert_x_kernel(const float* __restrict__ x,
                                 unsigned short* __restrict__ x_bf) {
    const int n = MROWS * D_DIM;
    for (int i = blockIdx.x * blockDim.x + threadIdx.x; i < n;
         i += gridDim.x * blockDim.x) {
        x_bf[i] = f2bf(x[i]);
    }
}

// ---------------------------------------------------------------------------
// Init recurrent state: h0 -> bf16 ping buffer, c0 -> f32 state buffer.
// ---------------------------------------------------------------------------
__global__ void init_state_kernel(const float* __restrict__ h0,
                                  const float* __restrict__ c0,
                                  unsigned short* __restrict__ h_bf,
                                  float* __restrict__ c_state) {
    const int n = B_DIM * H_DIM;
    for (int i = blockIdx.x * blockDim.x + threadIdx.x; i < n;
         i += gridDim.x * blockDim.x) {
        h_bf[i]    = f2bf(h0[i]);
        c_state[i] = c0[i];
    }
}

// ---------------------------------------------------------------------------
// Big parallel GEMM: Xproj[b*T+t, :] = x[b,t,:] @ Wi + bias.
// M=32768, N=2048, K=512. 8 waves/block, one 16x16 tile per wave.
// ---------------------------------------------------------------------------
__global__ __launch_bounds__(256) void xproj_kernel(
    const unsigned short* __restrict__ x_bf,  // [M, D] bf16
    const unsigned short* __restrict__ Wi,    // [D, 4H] bf16
    const float* __restrict__ bias,           // [4H]
    float* __restrict__ Xproj) {              // [M, 4H] f32
    const int tid   = threadIdx.x;
    const int lane  = tid & 31;
    const int wave  = tid >> 5;
    const int tile  = blockIdx.x * 8 + wave;   // 0 .. 262143
    const int ntile = tile & 127;
    const int mtile = tile >> 7;
    const int n0    = ntile * 16;
    const int half  = lane >> 4;
    const int mrow  = mtile * 16 + (lane & 15);

    v8f acc = {};
    const unsigned short* arow = x_bf + (size_t)mrow * D_DIM;
    for (int kk = 0; kk < 16; ++kk) {
        const int k0 = kk * 32;
        AFrag a;
        copy16B(&a.s[0], arow + k0 + half * 8);
        copy16B(&a.s[8], arow + k0 + 16 + half * 8);

        AFrag b;
        const unsigned short* wp = Wi + (size_t)k0 * Z_DIM + n0 + (lane & 15);
        for (int j = 0; j < 8; ++j) b.s[j]     = wp[(size_t)(half * 8 + j) * Z_DIM];
        for (int j = 0; j < 8; ++j) b.s[8 + j] = wp[(size_t)(16 + half * 8 + j) * Z_DIM];

        acc = __builtin_amdgcn_wmma_f32_16x16x32_bf16(
            false, a.v, false, b.v, (short)0, acc, false, false);
    }

    const int nc = n0 + (lane & 15);
    const float bb = bias[nc];
#pragma unroll
    for (int r = 0; r < 8; ++r)
        Xproj[(size_t)(mtile * 16 + half * 8 + r) * Z_DIM + nc] = acc[r] + bb;
}

// ---------------------------------------------------------------------------
// One LSTM time step.
// grid = 32 workgroups (one per 16-wide H slab), block = 512 threads = 16 waves.
// MODE 0: fused x@Wi (f32 -> bf16 on the fly) + h@Wh     (32 WMMA/wave)
// MODE 1: fused x@Wi (pre-converted bf16)    + h@Wh     (32 WMMA/wave)
// MODE 2: Xproj precomputed; only h@Wh                  (16 WMMA/wave)
// ---------------------------------------------------------------------------
template <int MODE>
__global__ __launch_bounds__(512) void lstm_step_kernel(
    const float* __restrict__ x,                // [B, T, D] fp32
    const unsigned short* __restrict__ x_bf,    // [B, T, D] bf16 (MODE 1)
    const float* __restrict__ Xproj,            // [B*T, 4H] f32  (MODE 2)
    const int* __restrict__ lengths,            // [B]
    const int* __restrict__ rev_p,              // [1]
    const unsigned short* __restrict__ Wi,      // [D, 4H] bf16
    const unsigned short* __restrict__ Wh,      // [H, 4H] bf16
    const float* __restrict__ bias,             // [4H]
    const unsigned short* __restrict__ h_prev,  // [B, H] bf16
    float* __restrict__ c_state,                // [B, H] f32 (slab-disjoint)
    unsigned short* __restrict__ h_cur,         // [B, H] bf16 (out)
    float* __restrict__ out,                    // outputs ++ c_final ++ h_final
    int t) {
    __shared__ float zsh[4][B_DIM][16];

    const int tid  = threadIdx.x;
    const int lane = tid & 31;
    const int wave = tid >> 5;
    const int mt   = wave & 3;   // batch tile: rows [16mt, 16mt+16)
    const int g    = wave >> 2;  // gate 0..3 (i, f, g, o)
    const int slab = blockIdx.x; // H columns [16*slab, 16*slab+16)
    const int n0   = slab * 16 + g * H_DIM;
    const int half = lane >> 4;
    const int mrow = mt * 16 + (lane & 15);

    const int rev = rev_p[0];

    v8f acc = {};

    if constexpr (MODE != 2) {
        // ---- x_t @ Wi (K = 0..511) ----
        const int len   = lengths[mrow];
        const int src_t = rev ? (T_LEN - 1 - t + len) % T_LEN : t;
        for (int kk = 0; kk < 16; ++kk) {
            const int k0 = kk * 32;
            AFrag a;
            if constexpr (MODE == 0) {
                const float* xrow = x + ((size_t)mrow * T_LEN + src_t) * D_DIM;
#pragma unroll
                for (int j = 0; j < 8; ++j) a.s[j]     = f2bf(xrow[k0 + half * 8 + j]);
#pragma unroll
                for (int j = 0; j < 8; ++j) a.s[8 + j] = f2bf(xrow[k0 + 16 + half * 8 + j]);
            } else {
                const unsigned short* xrow =
                    x_bf + ((size_t)mrow * T_LEN + src_t) * D_DIM;
                copy16B(&a.s[0], xrow + k0 + half * 8);
                copy16B(&a.s[8], xrow + k0 + 16 + half * 8);
            }
            AFrag b;
            const unsigned short* wp = Wi + (size_t)k0 * Z_DIM + n0 + (lane & 15);
            for (int j = 0; j < 8; ++j) b.s[j]     = wp[(size_t)(half * 8 + j) * Z_DIM];
            for (int j = 0; j < 8; ++j) b.s[8 + j] = wp[(size_t)(16 + half * 8 + j) * Z_DIM];
            if (kk < 15) __builtin_prefetch(wp + (size_t)32 * Z_DIM, 0, 3);

            acc = __builtin_amdgcn_wmma_f32_16x16x32_bf16(
                false, a.v, false, b.v, (short)0, acc, false, false);
        }
    }

    // ---- h_{t-1} @ Wh (K = 0..511) ----
    const unsigned short* hrow = h_prev + (size_t)mrow * H_DIM;
    for (int kk = 0; kk < 16; ++kk) {
        const int k0 = kk * 32;
        AFrag a;
        copy16B(&a.s[0], hrow + k0 + half * 8);
        copy16B(&a.s[8], hrow + k0 + 16 + half * 8);

        AFrag b;
        const unsigned short* wp = Wh + (size_t)k0 * Z_DIM + n0 + (lane & 15);
        for (int j = 0; j < 8; ++j) b.s[j]     = wp[(size_t)(half * 8 + j) * Z_DIM];
        for (int j = 0; j < 8; ++j) b.s[8 + j] = wp[(size_t)(16 + half * 8 + j) * Z_DIM];
        if (kk < 15) __builtin_prefetch(wp + (size_t)32 * Z_DIM, 0, 3);

        acc = __builtin_amdgcn_wmma_f32_16x16x32_bf16(
            false, a.v, false, b.v, (short)0, acc, false, false);
    }

    // C/D layout: lane -> N = lane&15, VGPR r -> M = r + 8*(lane>>4)
#pragma unroll
    for (int r = 0; r < 8; ++r)
        zsh[g][mt * 16 + half * 8 + r][lane & 15] = acc[r];
    __syncthreads();

    // ---- gates: 1024 elements (64 rows x 16 cols), 2 per thread ----
    for (int e = tid; e < B_DIM * 16; e += 512) {
        const int row = e >> 4;
        const int col = e & 15;
        const int cg  = slab * 16 + col;  // column within H

        const int lrow = lengths[row];
        // flip is an involution: scan-step t lands at original position p(t)
        const int pt = rev ? (T_LEN - 1 - t + lrow) % T_LEN : t;

        float zi, zf, zg, zo;
        if constexpr (MODE == 2) {
            const float* Xq = Xproj + ((size_t)row * T_LEN + pt) * Z_DIM;
            zi = zsh[0][row][col] + Xq[cg];            // bias folded into Xproj
            zf = zsh[1][row][col] + Xq[H_DIM + cg];
            zg = zsh[2][row][col] + Xq[2 * H_DIM + cg];
            zo = zsh[3][row][col] + Xq[3 * H_DIM + cg];
        } else {
            zi = zsh[0][row][col] + bias[cg];
            zf = zsh[1][row][col] + bias[H_DIM + cg];
            zg = zsh[2][row][col] + bias[2 * H_DIM + cg];
            zo = zsh[3][row][col] + bias[3 * H_DIM + cg];
        }

        const size_t ci = (size_t)row * H_DIM + cg;
        const float cp = c_state[ci];
        const float cn = sigmoidf_(zf) * cp + sigmoidf_(zi) * tanhf(zg);
        const float hn = sigmoidf_(zo) * tanhf(cn);

        c_state[ci] = cn;
        h_cur[ci]   = f2bf(hn);

        out[((size_t)row * T_LEN + pt) * H_DIM + cg] = hn;

        if (t == lrow - 1) {
            float* fin = out + (size_t)B_DIM * T_LEN * H_DIM;
            fin[ci] = cn;                              // c_final
            fin[(size_t)B_DIM * H_DIM + ci] = hn;      // h_final
        }
    }
}

// ---------------------------------------------------------------------------
extern "C" void kernel_launch(void* const* d_in, const int* in_sizes, int n_in,
                              void* d_out, int out_size, void* d_ws, size_t ws_size,
                              hipStream_t stream) {
    const float* x       = (const float*)d_in[0];
    const int*   lengths = (const int*)d_in[1];
    const float* c0      = (const float*)d_in[2];
    const float* h0      = (const float*)d_in[3];
    const float* Wi      = (const float*)d_in[4];
    const float* Wh      = (const float*)d_in[5];
    const float* bias    = (const float*)d_in[6];
    const int*   rev     = (const int*)d_in[7];

    char* ws = (char*)d_ws;
    // workspace layout:
    //   [0,      2MB)   Wi_bf
    //   [2MB,    4MB)   Wh_bf
    //   [4MB,   +64KB)  hb0
    //   [...,   +64KB)  hb1
    //   [...,  +128KB)  c_state
    //   [4.25MB, 36.25MB) x_bf      (MODE >= 1)
    //   [40MB,  296MB)  Xproj f32   (MODE == 2)
    const size_t MB = 1024 * 1024;
    unsigned short* Wi_bf   = (unsigned short*)(ws);
    unsigned short* Wh_bf   = (unsigned short*)(ws + 2 * MB);
    unsigned short* hb0     = (unsigned short*)(ws + 4 * MB);
    unsigned short* hb1     = (unsigned short*)(ws + 4 * MB + 65536);
    float*          c_state = (float*)(ws + 4 * MB + 131072);
    unsigned short* x_bf    = (unsigned short*)(ws + 4 * MB + 262144);
    float*          Xproj   = (float*)(ws + 40 * MB);

    const size_t needB = 4 * MB + 262144 + (size_t)MROWS * D_DIM * 2;  // ~36.3 MB
    const size_t needA = 40 * MB + (size_t)MROWS * Z_DIM * 4;          // ~296 MB
    const int mode = (ws_size >= needA) ? 2 : (ws_size >= needB ? 1 : 0);

    float* out = (float*)d_out;

    convert_weights_kernel<<<1024, 256, 0, stream>>>(Wi, Wh, Wi_bf, Wh_bf);
    init_state_kernel<<<128, 256, 0, stream>>>(h0, c0, hb0, c_state);
    if (mode >= 1) {
        convert_x_kernel<<<4096, 256, 0, stream>>>(x, x_bf);
    }
    if (mode == 2) {
        // 262144 tiles / 8 waves per block
        xproj_kernel<<<32768, 256, 0, stream>>>(x_bf, Wi_bf, bias, Xproj);
    }

    for (int t = 0; t < T_LEN; ++t) {
        unsigned short* hp = (t & 1) ? hb1 : hb0;
        unsigned short* hc = (t & 1) ? hb0 : hb1;
        if (mode == 2) {
            lstm_step_kernel<2><<<H_DIM / 16, 512, 0, stream>>>(
                x, x_bf, Xproj, lengths, rev, Wi_bf, Wh_bf, bias, hp, c_state, hc, out, t);
        } else if (mode == 1) {
            lstm_step_kernel<1><<<H_DIM / 16, 512, 0, stream>>>(
                x, x_bf, Xproj, lengths, rev, Wi_bf, Wh_bf, bias, hp, c_state, hc, out, t);
        } else {
            lstm_step_kernel<0><<<H_DIM / 16, 512, 0, stream>>>(
                x, x_bf, Xproj, lengths, rev, Wi_bf, Wh_bf, bias, hp, c_state, hc, out, t);
        }
    }
}
